// TCFPLinear_18949395710713
// MI455X (gfx1250) — 8-bit matrix operand order repaired and checked by reading, compiled, not run on hardware
//
#include <hip/hip_runtime.h>

typedef __attribute__((ext_vector_type(16))) int   v16i;
typedef __attribute__((ext_vector_type(8)))  float v8f;
typedef __attribute__((ext_vector_type(4)))  int   v4i;
typedef __attribute__((ext_vector_type(2)))  int   v2i;
typedef __attribute__((ext_vector_type(4)))  float v4f;

__device__ __forceinline__ unsigned char f32_to_e4m3(float x) {
  unsigned int u = __float_as_uint(x);
  unsigned char s = (unsigned char)((u >> 24) & 0x80);
  float ax = fabsf(x);
  if (!(ax > 0.0f)) return s;
  ax = fminf(ax, 448.0f);
  int e = (int)((__float_as_uint(ax) >> 23) & 0xFF) - 127;
  if (e < -6) e = -6;
  float quant = __uint_as_float((unsigned)(e - 3 + 127) << 23);
  int iv = (int)rintf(ax / quant);
  int E, Mn;
  if (e == -6 && iv < 8) {
    E = 0; Mn = iv;
  } else {
    if (iv >= 16) { iv >>= 1; e += 1; }
    E = e + 7; Mn = iv - 8;
    if (E > 15 || (E == 15 && Mn > 6)) { E = 15; Mn = 6; }
  }
  return (unsigned char)(s | (E << 3) | Mn);
}

__device__ __forceinline__ float e4m3_to_f32(unsigned char c) {
  int E = (c >> 3) & 0xF, Mn = c & 7;
  float v;
  if (E == 0) v = (float)Mn * 0x1p-9f;
  else        v = __uint_as_float(((unsigned)(E + 120) << 23) | ((unsigned)Mn << 20));
  return (c & 0x80) ? -v : v;
}

__device__ __forceinline__ float wave_amax(float a) {
  #pragma unroll
  for (int off = 16; off >= 1; off >>= 1)
    a = fmaxf(a, __shfl_xor(a, off, 32));
  return a;
}

__global__ __launch_bounds__(256) void tcfp_quant_x(
    const float* __restrict__ x, unsigned char* __restrict__ qx,
    float* __restrict__ invsx, int M) {
  constexpr int K = 1024, BS = 128;
  int wid = threadIdx.x >> 5, lane = threadIdx.x & 31;
  int row = blockIdx.x;
  int kb  = wid;
  const float* src = x + (size_t)row * K + kb * BS + lane * 4;
  float4 v = *(const float4*)src;
  float amax = fmaxf(fmaxf(fabsf(v.x), fabsf(v.y)), fmaxf(fabsf(v.z), fabsf(v.w)));
  amax = fmaxf(wave_amax(amax), 1e-12f);
  float sc = 448.0f / amax;
  unsigned int pk = (unsigned)f32_to_e4m3(v.x * sc)
                  | ((unsigned)f32_to_e4m3(v.y * sc) << 8)
                  | ((unsigned)f32_to_e4m3(v.z * sc) << 16)
                  | ((unsigned)f32_to_e4m3(v.w * sc) << 24);
  *(unsigned int*)(qx + (size_t)row * K + kb * BS + lane * 4) = pk;
  if (lane == 0) invsx[(size_t)kb * M + row] = amax / 448.0f;
}

__global__ __launch_bounds__(256) void tcfp_quant_w(
    const float* __restrict__ w, unsigned char* __restrict__ qwh,
    unsigned char* __restrict__ qwl, float* __restrict__ invsh,
    float* __restrict__ invsl, int N) {
  constexpr int K = 1024, BS = 128;
  int wid = threadIdx.x >> 5, lane = threadIdx.x & 31;
  int n  = blockIdx.x;
  int kb = wid;
  const float* src = w + (size_t)n * K + kb * BS + lane * 4;
  float4 v = *(const float4*)src;

  float amax1 = fmaxf(fmaxf(fabsf(v.x), fabsf(v.y)), fmaxf(fabsf(v.z), fabsf(v.w)));
  amax1 = fmaxf(wave_amax(amax1), 1e-12f);
  float s1 = 448.0f / amax1, inv1 = amax1 / 448.0f;
  unsigned char h0 = f32_to_e4m3(v.x * s1), h1 = f32_to_e4m3(v.y * s1);
  unsigned char h2 = f32_to_e4m3(v.z * s1), h3 = f32_to_e4m3(v.w * s1);
  unsigned int pkh = (unsigned)h0 | ((unsigned)h1 << 8) | ((unsigned)h2 << 16) | ((unsigned)h3 << 24);

  float r0 = v.x - e4m3_to_f32(h0) * inv1;
  float r1 = v.y - e4m3_to_f32(h1) * inv1;
  float r2 = v.z - e4m3_to_f32(h2) * inv1;
  float r3 = v.w - e4m3_to_f32(h3) * inv1;
  float amax2 = fmaxf(fmaxf(fabsf(r0), fabsf(r1)), fmaxf(fabsf(r2), fabsf(r3)));
  amax2 = fmaxf(wave_amax(amax2), 1e-12f);
  float s2 = 448.0f / amax2;
  unsigned int pkl = (unsigned)f32_to_e4m3(r0 * s2)
                   | ((unsigned)f32_to_e4m3(r1 * s2) << 8)
                   | ((unsigned)f32_to_e4m3(r2 * s2) << 16)
                   | ((unsigned)f32_to_e4m3(r3 * s2) << 24);

  size_t boff = (size_t)n * K + kb * BS + lane * 4;
  *(unsigned int*)(qwh + boff) = pkh;
  *(unsigned int*)(qwl + boff) = pkl;
  if (lane == 0) {
    invsh[(size_t)kb * N + n] = inv1;
    invsl[(size_t)kb * N + n] = amax2 / 448.0f;
  }
}

__global__ __launch_bounds__(256) void tcfp_gemm_fp8(
    const unsigned char* __restrict__ qx, const unsigned char* __restrict__ qwh,
    const unsigned char* __restrict__ qwl, const float* __restrict__ invsx,
    const float* __restrict__ invsh, const float* __restrict__ invsl,
    const float* __restrict__ bias, float* __restrict__ out, int M) {
  constexpr int K = 1024, N = 1024, NB = 8;
  int wid  = threadIdx.x >> 5;
  int lane = threadIdx.x & 31;
  int lrow = lane & 15;
  int lhi  = lane >> 4;
  int m0 = blockIdx.x * 32 + (wid >> 2) * 16;
  int n0 = blockIdx.y * 64 + (wid & 3) * 16;

  const unsigned char* Ab = qx  + (size_t)(m0 + lrow) * K;
  const unsigned char* Hb = qwh + (size_t)(n0 + lrow) * K;
  const unsigned char* Lb = qwl + (size_t)(n0 + lrow) * K;

  v8f acc = {};
  for (int kb = 0; kb < NB; ++kb) {
    int kbase = kb * 128;
    v16i a;
    #pragma unroll
    for (int t = 0; t < 8; ++t) {
      int off = kbase + (t >> 1) * 32 + lhi * 16 + (t & 1) * 8;
      v2i p = *(const v2i*)(Ab + off);
      a[2 * t] = p.x; a[2 * t + 1] = p.y;
    }
    v16i bh, bl;
    #pragma unroll
    for (int q = 0; q < 4; ++q) {
      int off = kbase + q * 32 + lhi * 16;
      v4i h = *(const v4i*)(Hb + off);
      v4i l = *(const v4i*)(Lb + off);
      bh[4 * q] = h.x; bh[4 * q + 1] = h.y; bh[4 * q + 2] = h.z; bh[4 * q + 3] = h.w;
      bl[4 * q] = l.x; bl[4 * q + 1] = l.y; bl[4 * q + 2] = l.z; bl[4 * q + 3] = l.w;
    }

    v8f z = {};
    v8f w0 = __builtin_amdgcn_wmma_f32_16x16x128_fp8_fp8(a, bh, (short)0, z, false, false);
    v8f w1 = __builtin_amdgcn_wmma_f32_16x16x128_fp8_fp8(a, bl, (short)0, z, false, false);

    float ibh = invsh[(size_t)kb * N + n0 + lrow];
    float ibl = invsl[(size_t)kb * N + n0 + lrow];
    const float* iap = invsx + (size_t)kb * M + m0 + lhi * 8;
    v4f ia0 = *(const v4f*)iap;
    v4f ia1 = *(const v4f*)(iap + 4);
    #pragma unroll
    for (int r = 0; r < 8; ++r) {
      float iar = (r < 4) ? ia0[r & 3] : ia1[r & 3];
      acc[r] += (w0[r] * ibh + w1[r] * ibl) * iar;
    }
  }

  float b = bias[n0 + lrow];
  #pragma unroll
  for (int r = 0; r < 8; ++r) {
    out[(size_t)(m0 + lhi * 8 + r) * N + n0 + lrow] = acc[r] + b;
  }
}

extern "C" void kernel_launch(void* const* d_in, const int* in_sizes, int n_in,
                              void* d_out, int out_size, void* d_ws, size_t ws_size,
                              hipStream_t stream) {
  (void)n_in; (void)out_size; (void)ws_size;
  const float* x    = (const float*)d_in[0];
  const float* w    = (const float*)d_in[1];
  const float* bias = (const float*)d_in[2];
  float* out = (float*)d_out;

  constexpr int K = 1024, N = 1024, NB = 8;
  const int M = in_sizes[0] / K;

  char* ws = (char*)d_ws;
  unsigned char* qx    = (unsigned char*)ws;
  unsigned char* qwh   = qx  + (size_t)M * K;
  unsigned char* qwl   = qwh + (size_t)N * K;
  float*         invsx = (float*)(qwl + (size_t)N * K);
  float*         invsh = invsx + (size_t)NB * M;
  float*         invsl = invsh + (size_t)NB * N;

  tcfp_quant_w<<<N, 256, 0, stream>>>(w, qwh, qwl, invsh, invsl, N);
  tcfp_quant_x<<<M, 256, 0, stream>>>(x, qx, invsx, M);

  dim3 grid(M / 32, N / 64);
  tcfp_gemm_fp8<<<grid, 256, 0, stream>>>(qx, qwh, qwl, invsx, invsh, invsl,
                                          bias, out, M);
}
